// SimpleMoEModel_2834678415768
// MI455X (gfx1250) — matrix operand order repaired and checked by reading, not compiled, not run on hardware
//
#include <hip/hip_runtime.h>
#include <hip/hip_bf16.h>
#include <math.h>

// ---------------------------------------------------------------------------
// Types for CDNA5 WMMA (wave32): v16bf = 8 VGPRs of bf16 pairs, v8f = 8 VGPRs.
// ---------------------------------------------------------------------------
typedef __attribute__((ext_vector_type(16))) __bf16 v16bf;
typedef __attribute__((ext_vector_type(8)))  __bf16 v8bf;
typedef __attribute__((ext_vector_type(8)))  float  v8f;
typedef unsigned int u32;
typedef __attribute__((ext_vector_type(4))) u32 v4u;
typedef __attribute__((ext_vector_type(8))) int v8i;
typedef __attribute__((ext_vector_type(4))) int v4i;

// Problem dims (fixed by reference)
#define BB 8
#define SS 512
#define DD 1024
#define EE 8
#define FF 4096
#define MM (BB * SS)   // 4096 tokens

// GEMM tiling
#define KCHUNK 256     // K elements staged per TDM transfer
#define BLK_N  128     // block tile N (cols)
#define BLK_M  128     // block tile M (rows)

// ---------------------------------------------------------------------------
// Elementwise f32 -> bf16 cast
// ---------------------------------------------------------------------------
__global__ void cast_bf16_kernel(const float* __restrict__ src,
                                 __bf16* __restrict__ dst, size_t n) {
  size_t i = (size_t)blockIdx.x * blockDim.x + threadIdx.x;
  size_t stride = (size_t)gridDim.x * blockDim.x;
  for (; i < n; i += stride) dst[i] = (__bf16)src[i];
}

// ---------------------------------------------------------------------------
// Tiled transpose + cast:  src[batch][R][C] f32  ->  dst[batch][C][R] bf16
// ---------------------------------------------------------------------------
__global__ void transpose_cast_kernel(const float* __restrict__ src,
                                      __bf16* __restrict__ dst,
                                      int R, int C) {
  __shared__ float tile[32][33];
  const size_t base = (size_t)blockIdx.z * R * C;
  const int c0 = blockIdx.x * 32;
  const int r0 = blockIdx.y * 32;
  const int tx = threadIdx.x, ty = threadIdx.y;

  #pragma unroll
  for (int j = 0; j < 4; ++j) {
    int r = r0 + ty + j * 8;
    tile[ty + j * 8][tx] = src[base + (size_t)r * C + (c0 + tx)];
  }
  __syncthreads();

  const size_t obase = (size_t)blockIdx.z * C * R;
  #pragma unroll
  for (int j = 0; j < 4; ++j) {
    int c = c0 + ty + j * 8;
    dst[obase + (size_t)c * R + (r0 + tx)] = (__bf16)tile[tx][ty + j * 8];
  }
}

// ---------------------------------------------------------------------------
// Zero init for the MoE accumulator + loss output.
// ---------------------------------------------------------------------------
__global__ void zero_kernel(float* __restrict__ p, size_t n, float* __restrict__ out) {
  size_t i = (size_t)blockIdx.x * blockDim.x + threadIdx.x;
  size_t stride = (size_t)gridDim.x * blockDim.x;
  for (; i < n; i += stride) p[i] = 0.0f;
  if (blockIdx.x == 0 && threadIdx.x == 0) *out = 0.0f;
}

// ---------------------------------------------------------------------------
// Issue one TDM transfer: Bt tile [BLK_N rows x KCHUNK cols] -> LDS.
//   Bt row-major [N x K] bf16; tile start at (rowBase, kBase).
//   LDS layout: [BLK_N][KCHUNK] bf16, contiguous (64 KB).
// D# packing per CDNA5 ISA (2D tensor, groups 2/3 zero).
// ---------------------------------------------------------------------------
__device__ __forceinline__ void tdm_load_b_tile(const __bf16* tileStart,
                                                u32 ldsByteAddr, int K) {
  const unsigned long long gaddr = (unsigned long long)(const void*)tileStart;
  // Group 0: count=1 | lds_addr | global_addr[56:0] | type=2
  v4u g0;
  g0.x = 1u;                                            // count = 1 valid descriptor
  g0.y = ldsByteAddr;                                   // lds_addr (bytes)
  g0.z = (u32)gaddr;                                    // global_addr[31:0]
  g0.w = ((u32)(gaddr >> 32) & 0x01FFFFFFu) | (2u << 30); // global_addr[56:32] | type=2
  // Group 1: data_size=1(2B); tensor_dim0=KCHUNK; tensor_dim1=BLK_N;
  //          tile_dim0=KCHUNK; tile_dim1=BLK_N; tensor_dim0_stride=K.
  v8i g1;
  g1[0] = (int)(1u << 16);                 // wg_mask=0 | data_size=1 (2 bytes)
  g1[1] = (int)((u32)KCHUNK << 16);        // abar_addr=0 | tensor_dim0[15:0]
  g1[2] = (int)((u32)BLK_N << 16);         // tensor_dim0[31:16]=0 | tensor_dim1[15:0]
  g1[3] = (int)((u32)KCHUNK << 16);        // tensor_dim1[31:16]=0 | tile_dim0
  g1[4] = (int)BLK_N;                      // tile_dim1 | tile_dim2=0
  g1[5] = K;                               // tensor_dim0_stride[31:0]
  g1[6] = 0;                               // stride hi | dim1_stride lo (unused, 2D)
  g1[7] = 0;
  v4i z4 = {0, 0, 0, 0};
#if __clang_major__ >= 23
  v8i z8 = {0, 0, 0, 0, 0, 0, 0, 0};
  __builtin_amdgcn_tensor_load_to_lds(g0, g1, z4, z4, z8, 0);
#else
  __builtin_amdgcn_tensor_load_to_lds(g0, g1, z4, z4, 0);
#endif
}

// ---------------------------------------------------------------------------
// BF16 WMMA GEMM:  C[M,N] = A[M,K] * Bt[N,K]^T  (+bias, +relu, +gated accum)
// Block = 128 threads = 4 waves (2x2); block tile 128x128; wave tile 64x64
// (16 accumulators, 4x4 v_wmma_f32_16x16x32_bf16 per 32-K step).
// B panel staged to LDS via Tensor Data Mover (TDM), KCHUNK at a time;
// A fragments streamed from global (L2-resident) with global_prefetch_b8.
// ---------------------------------------------------------------------------
template <bool RELU, bool WF32, bool WBF16, bool GATED>
__global__ __launch_bounds__(128) void
gemm_bf16_kernel(const __bf16* __restrict__ A,
                 const __bf16* __restrict__ Bt,
                 const float* __restrict__ bias,
                 float* __restrict__ Cf,
                 __bf16* __restrict__ Cb,
                 float* __restrict__ accum,
                 const float* __restrict__ gate,
                 int gateIdx,
                 int M, int N, int K) {
  __shared__ __bf16 ldsB[BLK_N * KCHUNK];   // 64 KB

  const int lane   = threadIdx.x & 31;
  const int wave   = threadIdx.x >> 5;      // 0..3
  const int laneLo = lane & 15;
  const int laneHi = (lane >> 4) & 1;

  const int rowBase  = blockIdx.y * BLK_M + (wave >> 1) * 64;
  const int colBase  = blockIdx.x * BLK_N + (wave & 1) * 64;
  const int colBlock = blockIdx.x * BLK_N;
  const int colLocal = (wave & 1) * 64;     // wave's col offset inside LDS panel

  v8f acc[4][4] = {};

  // Per-lane A base pointers (two 16B chunks per fragment: K lo-half / hi-half).
  const __bf16* aPtr[4];
  #pragma unroll
  for (int r = 0; r < 4; ++r)
    aPtr[r] = A + (size_t)(rowBase + r * 16 + laneLo) * K + laneHi * 16;

  // Flat LDS-aperture address truncates to the LDS byte offset (ISA 10.2).
  const u32 ldsByteAddr = (u32)(unsigned long long)(const void*)ldsB;

  for (int kc = 0; kc < K; kc += KCHUNK) {
    __syncthreads();                         // previous chunk fully consumed
    if (wave == 0) {
      tdm_load_b_tile(Bt + (size_t)colBlock * K + kc, ldsByteAddr, K);
      __builtin_amdgcn_s_wait_tensorcnt(0);  // TENSORcnt==0: tile resident in LDS
    }
    __syncthreads();                         // publish to all 4 waves

    for (int k0 = 0; k0 < KCHUNK; k0 += 32) {
      // A fragments (global, bf16): lanes 0-15 K=[0..15], 16-31 K=[16..31]
      v16bf a[4];
      #pragma unroll
      for (int r = 0; r < 4; ++r) {
        union { v16bf v; v8bf h[2]; } u;
        u.h[0] = *(const v8bf*)(aPtr[r] + kc + k0);
        u.h[1] = *(const v8bf*)(aPtr[r] + kc + k0 + 8);
        a[r] = u.v;
      }
      __builtin_prefetch((const void*)(aPtr[0] + kc + k0 + 32), 0, 1);
      __builtin_prefetch((const void*)(aPtr[2] + kc + k0 + 32), 0, 1);

      // B fragments (LDS): lane n<16 -> row colLocal+c*16+n, K=[0..15];
      //                    lane n>=16 -> same row set, K=[16..31]
      v16bf bfr[4];
      #pragma unroll
      for (int c = 0; c < 4; ++c)
        bfr[c] = *(const v16bf*)(ldsB + (colLocal + c * 16 + laneLo) * KCHUNK
                                      + k0 + laneHi * 16);

      #pragma unroll
      for (int r = 0; r < 4; ++r)
        #pragma unroll
        for (int c = 0; c < 4; ++c)
          acc[r][c] = __builtin_amdgcn_wmma_f32_16x16x32_bf16(
              false, a[r], false, bfr[c], (short)0, acc[r][c], false, false);
    }
  }

  // Epilogue. C/D layout: element v -> row = tileRow + laneHi*8 + v, col = tileCol + laneLo.
  #pragma unroll
  for (int r = 0; r < 4; ++r) {
    #pragma unroll
    for (int c = 0; c < 4; ++c) {
      const int col  = colBase + c * 16 + laneLo;
      const float bval = bias ? bias[col] : 0.0f;
      #pragma unroll
      for (int v = 0; v < 8; ++v) {
        const int row = rowBase + r * 16 + laneHi * 8 + v;
        float val = acc[r][c][v] + bval;
        if (RELU) val = fmaxf(val, 0.0f);
        const size_t idx = (size_t)row * N + col;
        if (WF32)  Cf[idx] = val;
        if (WBF16) Cb[idx] = (__bf16)val;
        if (GATED) {
          const float g = gate[(size_t)row * EE + gateIdx];
          accum[idx] += g * val;
        }
      }
    }
  }
}

// ---------------------------------------------------------------------------
// Gate: logits = lin @ Wg  [M,E]; softmax over E; top-2 renormalized -> gate[M,E]
// One wave per token. lane handles e = lane&7, K-partition = lane>>3.
// ---------------------------------------------------------------------------
__global__ void gate_kernel(const float* __restrict__ lin,
                            const float* __restrict__ Wg,
                            float* __restrict__ gate) {
  const int wave = threadIdx.x >> 5;
  const int lane = threadIdx.x & 31;
  const int m = blockIdx.x * 8 + wave;
  const int e = lane & 7;
  const int part = lane >> 3;       // 0..3

  const float* lrow = lin + (size_t)m * DD;
  float s = 0.0f;
  for (int k = part; k < DD; k += 4) s += lrow[k] * Wg[(size_t)k * EE + e];
  s += __shfl_xor(s, 8, 32);
  s += __shfl_xor(s, 16, 32);       // every lane now holds the full logit for its e

  float lg[EE];
  #pragma unroll
  for (int j = 0; j < EE; ++j) lg[j] = __shfl(s, j, 32);

  float m1 = lg[0]; int i1 = 0;
  #pragma unroll
  for (int j = 1; j < EE; ++j) if (lg[j] > m1) { m1 = lg[j]; i1 = j; }
  float m2 = -3.402823466e+38f; int i2 = 0;
  #pragma unroll
  for (int j = 0; j < EE; ++j)
    if (j != i1 && lg[j] > m2) { m2 = lg[j]; i2 = j; }

  const float p2 = __expf(m2 - m1);
  const float g1 = 1.0f / (1.0f + p2);
  const float g2 = p2 / (1.0f + p2);

  if (lane < EE)
    gate[(size_t)m * EE + lane] = (lane == i1) ? g1 : ((lane == i2) ? g2 : 0.0f);
}

// ---------------------------------------------------------------------------
// Final: sent_emb = mean_s(x + moe); log_softmax over D; CE at y[b]; mean over B.
// ---------------------------------------------------------------------------
__global__ void loss_kernel(const float* __restrict__ x,
                            const float* __restrict__ moe,
                            const int* __restrict__ y,
                            float* __restrict__ out) {
  __shared__ float red[256];
  const int b = blockIdx.x;
  const int t = threadIdx.x;

  float v[4];
  #pragma unroll
  for (int j = 0; j < 4; ++j) {
    const int d = t + j * 256;
    const float* xp = x   + (size_t)b * SS * DD + d;
    const float* mp = moe + (size_t)b * SS * DD + d;
    float s = 0.0f;
    for (int si = 0; si < SS; ++si) s += xp[(size_t)si * DD] + mp[(size_t)si * DD];
    v[j] = s * (1.0f / SS);
  }

  float mx = fmaxf(fmaxf(v[0], v[1]), fmaxf(v[2], v[3]));
  red[t] = mx;
  __syncthreads();
  for (int s = 128; s > 0; s >>= 1) {
    if (t < s) red[t] = fmaxf(red[t], red[t + s]);
    __syncthreads();
  }
  const float M = red[0];
  __syncthreads();

  float se = 0.0f;
  #pragma unroll
  for (int j = 0; j < 4; ++j) se += __expf(v[j] - M);
  red[t] = se;
  __syncthreads();
  for (int s = 128; s > 0; s >>= 1) {
    if (t < s) red[t] += red[t + s];
    __syncthreads();
  }
  const float logZ = logf(red[0]);

  const int yb = y[b];
  if ((yb & 255) == t) {
    const int j = yb >> 8;
    const float logp = v[j] - M - logZ;
    atomicAdd(out, -logp * (1.0f / BB));
  }
}

// ---------------------------------------------------------------------------
// Host-side orchestration
// ---------------------------------------------------------------------------
extern "C" void kernel_launch(void* const* d_in, const int* in_sizes, int n_in,
                              void* d_out, int out_size, void* d_ws, size_t ws_size,
                              hipStream_t stream) {
  (void)in_sizes; (void)n_in; (void)out_size; (void)ws_size;

  const float* x  = (const float*)d_in[0];   // [B,S,D]
  const int*   y  = (const int*)  d_in[1];   // [B]
  const float* W  = (const float*)d_in[2];   // [D,D]  (lin = x @ W^T + b)
  const float* b  = (const float*)d_in[3];   // [D]
  const float* Wg = (const float*)d_in[4];   // [D,E]
  const float* w1 = (const float*)d_in[5];   // [E,D,F]
  const float* b1 = (const float*)d_in[6];   // [E,F]
  const float* w2 = (const float*)d_in[7];   // [E,F,D]
  const float* b2 = (const float*)d_in[8];   // [E,D]
  float* out = (float*)d_out;

  // Workspace layout
  char* ws = (char*)d_ws;
  size_t off = 0;
  auto alloc = [&](size_t bytes) { char* p = ws + off; off += (bytes + 255) & ~(size_t)255; return p; };
  __bf16* x_bf   = (__bf16*)alloc((size_t)MM * DD * 2);      //  8.4 MB
  __bf16* W_bf   = (__bf16*)alloc((size_t)DD * DD * 2);      //  2.1 MB
  __bf16* w1T_bf = (__bf16*)alloc((size_t)EE * FF * DD * 2); // 67.1 MB  [E][F][D]
  __bf16* w2T_bf = (__bf16*)alloc((size_t)EE * DD * FF * 2); // 67.1 MB  [E][D][F]
  float*  lin_f  = (float*) alloc((size_t)MM * DD * 4);      // 16.8 MB
  __bf16* lin_bf = (__bf16*)alloc((size_t)MM * DD * 2);      //  8.4 MB
  __bf16* h_bf   = (__bf16*)alloc((size_t)MM * FF * 2);      // 33.6 MB (reused per expert)
  float*  gateP  = (float*) alloc((size_t)MM * EE * 4);      //  0.13 MB
  float*  moe    = (float*) alloc((size_t)MM * DD * 4);      // 16.8 MB

  // 0) zero moe accumulator + loss
  zero_kernel<<<2048, 256, 0, stream>>>(moe, (size_t)MM * DD, out);

  // 1) bf16 casts
  cast_bf16_kernel<<<4096, 256, 0, stream>>>(x, x_bf, (size_t)MM * DD);
  cast_bf16_kernel<<<1024, 256, 0, stream>>>(W, W_bf, (size_t)DD * DD);

  // 2) transpose+cast expert weights: w1 [E,D,F]->[E,F,D], w2 [E,F,D]->[E,D,F]
  transpose_cast_kernel<<<dim3(FF / 32, DD / 32, EE), dim3(32, 8), 0, stream>>>(w1, w1T_bf, DD, FF);
  transpose_cast_kernel<<<dim3(DD / 32, FF / 32, EE), dim3(32, 8), 0, stream>>>(w2, w2T_bf, FF, DD);

  // 3) lin = x @ W^T + b   (write f32 for gate + bf16 for expert GEMMs)
  gemm_bf16_kernel<false, true, true, false>
      <<<dim3(DD / BLK_N, MM / BLK_M), 128, 0, stream>>>(
          x_bf, W_bf, b, lin_f, lin_bf, nullptr, nullptr, 0, MM, DD, DD);

  // 4) gating: softmax + top-2 renorm -> gate[M,E]
  gate_kernel<<<MM / 8, 256, 0, stream>>>(lin_f, Wg, gateP);

  // 5) experts: h = relu(lin @ w1[e] + b1[e]);  moe += gate[:,e] * (h @ w2[e] + b2[e])
  for (int e = 0; e < EE; ++e) {
    gemm_bf16_kernel<true, false, true, false>
        <<<dim3(FF / BLK_N, MM / BLK_M), 128, 0, stream>>>(
            lin_bf, w1T_bf + (size_t)e * FF * DD, b1 + (size_t)e * FF,
            nullptr, h_bf, nullptr, nullptr, 0, MM, FF, DD);

    gemm_bf16_kernel<false, false, false, true>
        <<<dim3(DD / BLK_N, MM / BLK_M), 128, 0, stream>>>(
            h_bf, w2T_bf + (size_t)e * DD * FF, b2 + (size_t)e * DD,
            nullptr, nullptr, moe, gateP, e, MM, DD, FF);
  }

  // 6) residual + mean pool + log_softmax + CE
  loss_kernel<<<BB, 256, 0, stream>>>(x, moe, y, out);
}
